// MetricLearningLoss_35665408426430
// MI455X (gfx1250) — compile-verified, matrix operand order checked by reading, not run on hardware
//
#include <hip/hip_runtime.h>

typedef __attribute__((ext_vector_type(16))) __bf16 v16bf;
typedef __attribute__((ext_vector_type(8)))  float  v8f;

#define N_PTS 8192
#define DIMS  128
#define TILE64 64
#define T64 (N_PTS / TILE64)             // 128 tiles per dim
#define NTILES (T64 * (T64 + 1) / 2)     // 8256 triangular wave-tiles
#define WAVES_PER_BLOCK 8

// ---------------------------------------------------------------- sq norms
__global__ __launch_bounds__(256) void sqn_kernel(const float* __restrict__ x,
                                                  float* __restrict__ sqn) {
    int row = blockIdx.x * blockDim.x + threadIdx.x;
    if (row >= N_PTS) return;
    const float4* p = reinterpret_cast<const float4*>(x + (size_t)row * DIMS);
    float s = 0.f;
    #pragma unroll
    for (int i = 0; i < DIMS / 4; ++i) {
        float4 v = p[i];
        s += v.x * v.x + v.y * v.y + v.z * v.z + v.w * v.w;
    }
    sqn[row] = s;
}

// ------------------------------------------------------------ tile helpers
__device__ __forceinline__ int tri_start(int r) { return r * T64 - (r * (r - 1)) / 2; }

// Load 16 f32 for this lane's A/B fragment rows and split into hi/lo bf16.
// Element e of v16bf maps to K = base + 8*h + e (+8 extra for e>=8), which is
// exactly rows [off .. off+7] and [off+16 .. off+23] with off = 32*chunk + 8*h.
__device__ __forceinline__ void load_split(const float* __restrict__ row, int off,
                                           v16bf& hi, v16bf& lo) {
    const float4* p0 = reinterpret_cast<const float4*>(row + off);
    const float4* p1 = reinterpret_cast<const float4*>(row + off + 16);
    float v[16];
    float4 t0 = p0[0], t1 = p0[1], t2 = p1[0], t3 = p1[1];
    v[0]=t0.x; v[1]=t0.y; v[2]=t0.z;  v[3]=t0.w;
    v[4]=t1.x; v[5]=t1.y; v[6]=t1.z;  v[7]=t1.w;
    v[8]=t2.x; v[9]=t2.y; v[10]=t2.z; v[11]=t2.w;
    v[12]=t3.x; v[13]=t3.y; v[14]=t3.z; v[15]=t3.w;
    #pragma unroll
    for (int e = 0; e < 16; ++e) {
        __bf16 h = (__bf16)v[e];
        hi[e] = h;
        lo[e] = (__bf16)(v[e] - (float)h);
    }
}

// ----------------------------------------------- fused gram + loss per tile
__global__ __launch_bounds__(256) void
gram_tile_kernel(const float* __restrict__ x, const int* __restrict__ labels,
                 const float* __restrict__ sqn, float* __restrict__ partials) {
    const int lane = threadIdx.x & 31;
    const int wave = threadIdx.x >> 5;
    const int t = blockIdx.x * WAVES_PER_BLOCK + wave;
    if (t >= NTILES) return;

    // decode triangular index t -> (ti <= tj)
    float twoT1 = 2.0f * T64 + 1.0f;
    int ti = (int)((twoT1 - sqrtf(twoT1 * twoT1 - 8.0f * (float)t)) * 0.5f);
    if (ti < 0) ti = 0;
    if (ti > T64 - 1) ti = T64 - 1;
    while (ti + 1 <= T64 - 1 && tri_start(ti + 1) <= t) ++ti;
    while (ti > 0 && tri_start(ti) > t) --ti;
    const int tj = ti + (t - tri_start(ti));

    const int I = ti * TILE64;
    const int J = tj * TILE64;
    const int h  = lane >> 4;   // half-wave selects K-group / M-group
    const int n0 = lane & 15;   // row within 16 (A) / col within 16 (B)

    v8f acc[4][4];
    #pragma unroll
    for (int a = 0; a < 4; ++a)
        #pragma unroll
        for (int b = 0; b < 4; ++b)
            #pragma unroll
            for (int e = 0; e < 8; ++e) acc[a][b][e] = 0.0f;

    const float* rowA[4];
    const float* rowB[4];
    #pragma unroll
    for (int a = 0; a < 4; ++a) rowA[a] = x + (size_t)(I + 16 * a + n0) * DIMS;
    #pragma unroll
    for (int b = 0; b < 4; ++b) rowB[b] = x + (size_t)(J + 16 * b + n0) * DIMS;

    #pragma unroll
    for (int c0 = 0; c0 < 4; ++c0) {            // K chunks of 32
        const int off = 32 * c0 + 8 * h;
        v16bf ah[4], al[4];
        #pragma unroll
        for (int a = 0; a < 4; ++a) load_split(rowA[a], off, ah[a], al[a]);
        #pragma unroll
        for (int b = 0; b < 4; ++b) {
            v16bf bh, bl;
            load_split(rowB[b], off, bh, bl);
            #pragma unroll
            for (int a = 0; a < 4; ++a) {
                // split-bf16 fp32 emulation: hi*hi + hi*lo + lo*hi
                acc[a][b] = __builtin_amdgcn_wmma_f32_16x16x32_bf16(
                    false, ah[a], false, bh, (short)0, acc[a][b], false, false);
                acc[a][b] = __builtin_amdgcn_wmma_f32_16x16x32_bf16(
                    false, ah[a], false, bl, (short)0, acc[a][b], false, false);
                acc[a][b] = __builtin_amdgcn_wmma_f32_16x16x32_bf16(
                    false, al[a], false, bh, (short)0, acc[a][b], false, false);
            }
        }
    }

    // fused epilogue: d2 = |xi|^2 + |xj|^2 - 2*gram, clamp, label-weighted sum
    float snj[4]; int lj[4];
    #pragma unroll
    for (int b = 0; b < 4; ++b) {
        int jj = J + 16 * b + n0;
        snj[b] = sqn[jj];
        lj[b]  = labels[jj];
    }
    float part = 0.f;
    #pragma unroll
    for (int a = 0; a < 4; ++a) {
        #pragma unroll
        for (int r = 0; r < 8; ++r) {           // C/D: M = r + 8*h
            const int ii = I + 16 * a + 8 * h + r;
            const float sni = sqn[ii];
            const int  li  = labels[ii];
            #pragma unroll
            for (int b = 0; b < 4; ++b) {
                float d2 = sni + snj[b] - 2.0f * acc[a][b][r];
                d2 = fmaxf(d2, 0.0f);
                part += (li == lj[b]) ? (-6.25f * d2) : (0.25f * d2);
            }
        }
    }
    if (ti != tj) part *= 2.0f;                 // symmetric off-diagonal tiles

    #pragma unroll
    for (int s = 16; s > 0; s >>= 1) part += __shfl_xor(part, s, 32);
    if (lane == 0) partials[t] = part;
}

// ---------------------------------------------------- deterministic reduce
__global__ __launch_bounds__(256) void reduce_kernel(const float* __restrict__ partials,
                                                     float* __restrict__ out) {
    __shared__ float sm[256];
    float s = 0.f;
    for (int i = threadIdx.x; i < NTILES; i += 256) s += partials[i];
    sm[threadIdx.x] = s;
    __syncthreads();
    for (int stride = 128; stride > 0; stride >>= 1) {
        if (threadIdx.x < stride) sm[threadIdx.x] += sm[threadIdx.x + stride];
        __syncthreads();
    }
    if (threadIdx.x == 0) out[0] = sm[0];
}

extern "C" void kernel_launch(void* const* d_in, const int* in_sizes, int n_in,
                              void* d_out, int out_size, void* d_ws, size_t ws_size,
                              hipStream_t stream) {
    const float* x      = (const float*)d_in[0];
    const int*   labels = (const int*)d_in[1];

    float* sqn      = (float*)d_ws;          // N_PTS floats
    float* partials = sqn + N_PTS;           // NTILES floats

    sqn_kernel<<<N_PTS / 256, 256, 0, stream>>>(x, sqn);
    gram_tile_kernel<<<(NTILES + WAVES_PER_BLOCK - 1) / WAVES_PER_BLOCK, 256, 0, stream>>>(
        x, labels, sqn, partials);
    reduce_kernel<<<1, 256, 0, stream>>>(partials, (float*)d_out);
}
